// PhrasalScaledDotProductAttention_59313498358240
// MI455X (gfx1250) — matrix operand order repaired and checked by reading, not compiled, not run on hardware
//
#include <hip/hip_runtime.h>

// ---------------------------------------------------------------------------
// CDNA5 (gfx1250) wave32 WMMA bf16 pipeline for Phrasal Scaled-Dot Attention
// ---------------------------------------------------------------------------

typedef __attribute__((ext_vector_type(16))) __bf16 v16bf;
typedef __attribute__((ext_vector_type(8)))  float  v8f;

union FragU { uint4 u[2]; v16bf v; };

__device__ __forceinline__ unsigned short f2bf(float f) {
    unsigned int u = __float_as_uint(f);
    u += 0x7FFFu + ((u >> 16) & 1u);          // round-to-nearest-even
    return (unsigned short)(u >> 16);
}

__device__ __forceinline__ void storeC(float* p, float v) { *p = v; }
__device__ __forceinline__ void storeC(unsigned short* p, float v) { *p = f2bf(v); }

// Stage a ROWSx32 tile into LDS as bf16. f32 sources are converted on the fly.
template<int ROWS, typename T>
__device__ __forceinline__ void load_tile(unsigned short* __restrict__ sdst,
                                          const T* __restrict__ src,
                                          int colOff, int ld) {
    constexpr int E = ROWS * 32 / 256;        // elements per thread (8 or 16)
    int lin = threadIdx.x * E;
    int r = lin >> 5, c = lin & 31;
    if constexpr (sizeof(T) == 4) {           // f32 -> bf16 convert path
        const float* p = reinterpret_cast<const float*>(src) + (long)r * ld + colOff + c;
        unsigned short* d = sdst + r * 32 + c;
        #pragma unroll
        for (int e = 0; e < E; e += 4) {
            float4 v = *reinterpret_cast<const float4*>(p + e);
            d[e + 0] = f2bf(v.x); d[e + 1] = f2bf(v.y);
            d[e + 2] = f2bf(v.z); d[e + 3] = f2bf(v.w);
        }
    } else {                                  // bf16 straight copy (b128)
        const unsigned short* p = reinterpret_cast<const unsigned short*>(src)
                                  + (long)r * ld + colOff + c;
        uint4* d = reinterpret_cast<uint4*>(sdst + r * 32 + c);
        const uint4* s = reinterpret_cast<const uint4*>(p);
        #pragma unroll
        for (int e = 0; e < E / 8; ++e) d[e] = s[e];
    }
}

// ---------------------------------------------------------------------------
// Generic batched GEMM: C[m,n] = alpha * sum_k A[m,k]*B[n,k] (+bias[n])
// A: [M,K] row-major (K contiguous), B: [N,K] row-major (K contiguous).
// Per-z operand offset = (z/zInner)*outer + (z%zInner)*inner.
// Block = 256 threads = 8 wave32 waves in a 4x2 grid; each wave owns a
// (WMT*16) x (WNT*16) sub-tile built from v_wmma_f32_16x16x32_bf16.
// ---------------------------------------------------------------------------
template<typename TA, typename TB, typename TC, int WMT, int WNT>
__global__ __launch_bounds__(256)
void gemm_tn(const TA* __restrict__ A, long aOuter, long aInner,
             const TB* __restrict__ B, long bOuter, long bInner,
             TC* __restrict__ C, long cOuter, long cInner,
             int zInner, int lda, int ldb, int ldc,
             const float* __restrict__ bias, float alpha, int K)
{
    constexpr int BM = 4 * WMT * 16;          // 128
    constexpr int BN = 2 * WNT * 16;          // 128 or 64
    __shared__ unsigned short sA[2][BM * 32];
    __shared__ unsigned short sB[2][BN * 32];

    int z = blockIdx.z;
    long zA = (long)(z / zInner) * aOuter + (long)(z % zInner) * aInner;
    long zB = (long)(z / zInner) * bOuter + (long)(z % zInner) * bInner;
    long zC = (long)(z / zInner) * cOuter + (long)(z % zInner) * cInner;

    const TA* Ab = A + zA + (long)blockIdx.y * BM * lda;
    const TB* Bb = B + zB + (long)blockIdx.x * BN * ldb;
    TC*       Cb = C + zC + (long)blockIdx.y * BM * ldc + (long)blockIdx.x * BN;

    int lane = threadIdx.x & 31;
    int wave = threadIdx.x >> 5;
    int wm = wave & 3, wn = wave >> 2;
    int wmBase = wm * (WMT * 16), wnBase = wn * (WNT * 16);
    int lm = lane & 15, ks = lane >> 4;       // ISA A/B lane layout selectors

    v8f acc[WMT][WNT] = {};

    int nk = K >> 5;
    load_tile<BM>(sA[0], Ab, 0, lda);
    load_tile<BN>(sB[0], Bb, 0, ldb);
    __syncthreads();

    for (int kt = 0; kt < nk; ++kt) {
        int buf = kt & 1;
        if (kt + 1 < nk) {                    // stage next K-slab (double buffer)
            load_tile<BM>(sA[buf ^ 1], Ab, (kt + 1) * 32, lda);
            load_tile<BN>(sB[buf ^ 1], Bb, (kt + 1) * 32, ldb);
        }
        if (kt + 2 < nk)                      // gfx1250 global_prefetch_b8
            __builtin_prefetch(reinterpret_cast<const char*>(Ab)
                               + ((long)(threadIdx.x >> 1) * lda + (kt + 2) * 32)
                                 * (long)sizeof(TA), 0, 1);

        // A fragment: 16x32 bf16. Lane L (L<16): row M=L, K halves [0..7],[16..23];
        // lanes 16..31: same rows, K halves [8..15],[24..31].
        v16bf aF[WMT], bF[WNT];
        #pragma unroll
        for (int i = 0; i < WMT; ++i) {
            const unsigned short* p = &sA[buf][(wmBase + i * 16 + lm) * 32];
            FragU f;
            f.u[0] = *reinterpret_cast<const uint4*>(p + ks * 8);
            f.u[1] = *reinterpret_cast<const uint4*>(p + 16 + ks * 8);
            aF[i] = f.v;
        }
        // B fragment: 32x16 bf16. Lane L: column N=L&15, K halves [(L>>4)*8..+7],[16+(L>>4)*8..+7].
        #pragma unroll
        for (int j = 0; j < WNT; ++j) {
            const unsigned short* p = &sB[buf][(wnBase + j * 16 + lm) * 32];
            FragU f;
            f.u[0] = *reinterpret_cast<const uint4*>(p + ks * 8);
            f.u[1] = *reinterpret_cast<const uint4*>(p + 16 + ks * 8);
            bF[j] = f.v;
        }
        #pragma unroll
        for (int i = 0; i < WMT; ++i)
            #pragma unroll
            for (int j = 0; j < WNT; ++j)
                acc[i][j] = __builtin_amdgcn_wmma_f32_16x16x32_bf16(
                    false, aF[i], false, bF[j], (short)0, acc[i][j], false, false);
        __syncthreads();
    }

    // D layout: VGPR r, lane L -> M = r + 8*(L>>4), N = L&15
    #pragma unroll
    for (int i = 0; i < WMT; ++i)
        #pragma unroll
        for (int j = 0; j < WNT; ++j) {
            int n = wnBase + j * 16 + lm;
            float bvn = bias ? bias[(long)blockIdx.x * BN + n] : 0.0f;
            #pragma unroll
            for (int r = 0; r < 8; ++r) {
                int m = wmBase + i * 16 + r + 8 * ks;
                storeC(Cb + (long)m * ldc + n, acc[i][j][r] * alpha + bvn);
            }
        }
}

// ---------------------------------------------------------------------------
// att[row,:] = softmax(att[row,:] + lambda*log(P[b,q,:]+eps)); row = (b,h,q)
// ---------------------------------------------------------------------------
__global__ __launch_bounds__(256)
void softmax_phrasal(float* __restrict__ att, const float* __restrict__ P,
                     const float* __restrict__ lambda_p)
{
    long row = blockIdx.x;                    // 0 .. B*H*N-1
    int b = (int)(row >> 14);                 // / (16*1024)
    int q = (int)(row & 1023);
    float* s = att + row * 1024;
    const float* p = P + ((long)b * 1024 + q) * 1024;
    float lam = lambda_p[0];

    __shared__ float red[8];
    int t = threadIdx.x, lane = t & 31, wave = t >> 5;

    float loc[4];
    float lmax = -3.4e38f;
    #pragma unroll
    for (int i = 0; i < 4; ++i) {
        int k = t + i * 256;
        float v = s[k] + lam * __logf(p[k] + 1e-9f);
        loc[i] = v;
        lmax = fmaxf(lmax, v);
    }
    #pragma unroll
    for (int off = 16; off > 0; off >>= 1)
        lmax = fmaxf(lmax, __shfl_xor(lmax, off, 32));
    if (lane == 0) red[wave] = lmax;
    __syncthreads();
    float rmax = red[0];
    #pragma unroll
    for (int w = 1; w < 8; ++w) rmax = fmaxf(rmax, red[w]);
    __syncthreads();

    float lsum = 0.f;
    #pragma unroll
    for (int i = 0; i < 4; ++i) { loc[i] = __expf(loc[i] - rmax); lsum += loc[i]; }
    #pragma unroll
    for (int off = 16; off > 0; off >>= 1)
        lsum += __shfl_xor(lsum, off, 32);
    if (lane == 0) red[wave] = lsum;
    __syncthreads();
    float rsum = 0.f;
    #pragma unroll
    for (int w = 0; w < 8; ++w) rsum += red[w];
    float inv = 1.0f / rsum;
    #pragma unroll
    for (int i = 0; i < 4; ++i) s[t + i * 256] = loc[i] * inv;
}

__global__ __launch_bounds__(256)
void sigmoid_inplace(float* __restrict__ P, const float* __restrict__ bb) {
    long i = (long)blockIdx.x * 256 + threadIdx.x;
    float v = P[i] + bb[0];
    P[i] = 1.0f / (1.0f + __expf(-v));
}

__global__ __launch_bounds__(256)
void transpose_f32(const float* __restrict__ in, float* __restrict__ out, int n) {
    __shared__ float tile[32][33];
    int bx = blockIdx.x * 32, by = blockIdx.y * 32;
    int tx = threadIdx.x & 31, r0 = (threadIdx.x >> 5) * 4;
    #pragma unroll
    for (int r = 0; r < 4; ++r)
        tile[r0 + r][tx] = in[(long)(by + r0 + r) * n + bx + tx];
    __syncthreads();
    #pragma unroll
    for (int r = 0; r < 4; ++r)
        out[(long)(bx + r0 + r) * n + by + tx] = tile[tx][r0 + r];
}

__global__ __launch_bounds__(256)
void transpose_bf16(const unsigned short* __restrict__ in,
                    unsigned short* __restrict__ out, int n) {
    __shared__ unsigned short tile[32][33];
    long zo = (long)blockIdx.z * n * n;
    int bx = blockIdx.x * 32, by = blockIdx.y * 32;
    int tx = threadIdx.x & 31, r0 = (threadIdx.x >> 5) * 4;
    #pragma unroll
    for (int r = 0; r < 4; ++r)
        tile[r0 + r][tx] = in[zo + (long)(by + r0 + r) * n + bx + tx];
    __syncthreads();
    #pragma unroll
    for (int r = 0; r < 4; ++r)
        out[zo + (long)(bx + r0 + r) * n + by + tx] = tile[tx][r0 + r];
}

// ---------------------------------------------------------------------------
extern "C" void kernel_launch(void* const* d_in, const int* in_sizes, int n_in,
                              void* d_out, int out_size, void* d_ws, size_t ws_size,
                              hipStream_t stream)
{
    (void)in_sizes; (void)n_in; (void)out_size; (void)ws_size;

    const float* queries = (const float*)d_in[0];
    const float* keys    = (const float*)d_in[1];
    const float* values  = (const float*)d_in[2];
    const float* Wq = (const float*)d_in[3];
    const float* bq = (const float*)d_in[4];
    const float* Wk = (const float*)d_in[5];
    const float* bk = (const float*)d_in[6];
    const float* Wv = (const float*)d_in[7];
    const float* bv = (const float*)d_in[8];
    const float* Wo = (const float*)d_in[9];
    const float* bo = (const float*)d_in[10];
    const float* Wb = (const float*)d_in[11];
    const float* bb = (const float*)d_in[12];
    const float* lam = (const float*)d_in[13];

    constexpr long outElems = 8L * 1024 * 1024;        // 8,388,608  (B*N*Dm)
    constexpr long attElems = 8L * 16 * 1024 * 1024;   // 134,217,728

    float* out = (float*)d_out;
    float* att = out + outElems;     // also used as raw-score scratch
    float* P   = att + attElems;     // also used as raw bilinear-score scratch

    // bf16 workspace (all fully written before read; ~100 MB total)
    unsigned short* Qp = (unsigned short*)d_ws;        // [B,N,H*Dk]
    unsigned short* Kp = Qp + outElems;
    unsigned short* Vp = Kp + outElems;
    unsigned short* xW = Vp + outElems;                // [B,N,Dm]
    unsigned short* Vt = xW + outElems;                // [B,H,Dv,N]
    unsigned short* Ob = Vt + outElems;                // [B,N,H*Dv]
    float* WbT = (float*)(Ob + outElems);              // [Dm,Dm] f32

    dim3 blk(256);

    // 0. WbT = Wb^T (so xW GEMM reads K-contiguous B operand)
    transpose_f32<<<dim3(32, 32), blk, 0, stream>>>(Wb, WbT, 1024);

    // 1-4. Projections + xW: [8192,1024] @ [1024,1024]^T, bf16 out
    gemm_tn<float, float, unsigned short, 2, 4><<<dim3(8, 64, 1), blk, 0, stream>>>(
        queries, 0, 0, Wq, 0, 0, Qp, 0, 0, 1, 1024, 1024, 1024, bq, 1.0f, 1024);
    gemm_tn<float, float, unsigned short, 2, 4><<<dim3(8, 64, 1), blk, 0, stream>>>(
        keys, 0, 0, Wk, 0, 0, Kp, 0, 0, 1, 1024, 1024, 1024, bk, 1.0f, 1024);
    gemm_tn<float, float, unsigned short, 2, 4><<<dim3(8, 64, 1), blk, 0, stream>>>(
        values, 0, 0, Wv, 0, 0, Vp, 0, 0, 1, 1024, 1024, 1024, bv, 1.0f, 1024);
    gemm_tn<float, float, unsigned short, 2, 4><<<dim3(8, 64, 1), blk, 0, stream>>>(
        queries, 0, 0, WbT, 0, 0, xW, 0, 0, 1, 1024, 1024, 1024, nullptr, 1.0f, 1024);

    // 5. Sp = xW @ queries^T  -> P region (raw), batched over b
    gemm_tn<unsigned short, float, float, 2, 4><<<dim3(8, 8, 8), blk, 0, stream>>>(
        xW, 1048576L, 0, queries, 1048576L, 0, P, 1048576L, 0, 1,
        1024, 1024, 1024, nullptr, 1.0f, 1024);

    // 6. P = sigmoid(Sp + bb)  (in place)
    sigmoid_inplace<<<dim3(32768), blk, 0, stream>>>(P, bb);

    // 7. S = (Q K^T)/sqrt(64) -> att region; z = b*16+h (zInner=16 strides)
    gemm_tn<unsigned short, unsigned short, float, 2, 4><<<dim3(8, 8, 128), blk, 0, stream>>>(
        Qp, 1048576L, 64L, Kp, 1048576L, 64L, att, 16777216L, 1048576L, 16,
        1024, 1024, 1024, nullptr, 0.125f, 64);

    // 8. att = softmax(S + lambda*log(P+eps))  (in place, one row per block)
    softmax_phrasal<<<dim3(131072), blk, 0, stream>>>(att, P, lam);

    // 9. Vt[b,h*64+d,n] = Vp[b,n,h*64+d]
    transpose_bf16<<<dim3(32, 32, 8), blk, 0, stream>>>(Vp, Vt, 1024);

    // 10. O = att @ V -> Ob laid out [b, q, h*64+d]  (N=64 tile config)
    gemm_tn<float, unsigned short, unsigned short, 2, 2><<<dim3(1, 8, 128), blk, 0, stream>>>(
        att, 16777216L, 1048576L, Vt, 1048576L, 65536L, Ob, 1048576L, 64L, 16,
        1024, 1024, 1024, nullptr, 1.0f, 1024);

    // 11. out = O @ Wo^T + bo
    gemm_tn<unsigned short, float, float, 2, 4><<<dim3(8, 64, 1), blk, 0, stream>>>(
        Ob, 0, 0, Wo, 0, 0, out, 0, 0, 1, 1024, 1024, 1024, bo, 1.0f, 1024);
}
